// SPELLB_26877905339110
// MI455X (gfx1250) — matrix operand order repaired and checked by reading, not compiled, not run on hardware
//
#include <hip/hip_runtime.h>
#include <hip/hip_bf16.h>

// ---------------------------------------------------------------------------
// Types for CDNA5 WMMA (wave32). 16x16x32 bf16 -> f32 accumulate.
// ---------------------------------------------------------------------------
typedef __attribute__((ext_vector_type(16))) __bf16 v16bf;
typedef __attribute__((ext_vector_type(8)))  __bf16 v8bf;
typedef __attribute__((ext_vector_type(8)))  float  v8f;

#define EPSBN 1e-5f

__device__ __forceinline__ v8f wmma_bf16(v16bf a, v16bf b, v8f c) {
  // (neg_a, A, neg_b, B, c_mod, C, reuse_a, reuse_b)
  return __builtin_amdgcn_wmma_f32_16x16x32_bf16(false, a, false, b, (short)0, c,
                                                 false, false);
}

// Monotonic uint key for float atomic max (handles negatives, -inf init).
__device__ __forceinline__ unsigned fkey(float f) {
  unsigned u = __float_as_uint(f);
  return (u & 0x80000000u) ? ~u : (u | 0x80000000u);
}
__device__ __forceinline__ float dekey(unsigned k) {
  unsigned u = (k & 0x80000000u) ? (k ^ 0x80000000u) : ~k;
  return __uint_as_float(u);
}
#define NEGINF_KEY 0x007FFFFFu  // fkey(-inf)

// A-fragment (16x32 bf16, M=row stripe) from row-major [*,64] bf16 activations.
// ISA layout: lane L: M=L%16, half=L/16; elems 0..7 -> K=half*8+0..7,
// elems 8..15 -> K=16+half*8+0..7 (within the 32-wide k-chunk).
__device__ __forceinline__ v16bf load_afrag(const __bf16* __restrict__ act,
                                            int row, int kc, int lane) {
  const int half = lane >> 4;
  const __bf16* p = act + (size_t)row * 64 + kc * 32;
  v8bf lo = *(const v8bf*)(p + half * 8);
  v8bf hi = *(const v8bf*)(p + 16 + half * 8);
  v16bf a;
#pragma unroll
  for (int i = 0; i < 8; ++i) { a[i] = lo[i]; a[8 + i] = hi[i]; }
  return a;
}

// Edge fragment: relu(Abuf[dst] + Bbuf[src]) in bf16.
__device__ __forceinline__ v16bf edge_frag(const __bf16* __restrict__ Ab,
                                           const __bf16* __restrict__ Bb,
                                           int dn, int sn, int kc, int lane) {
  const int half = lane >> 4;
  const __bf16* pa = Ab + (size_t)dn * 64 + kc * 32;
  const __bf16* pb = Bb + (size_t)sn * 64 + kc * 32;
  v8bf a0 = *(const v8bf*)(pa + half * 8);
  v8bf a1 = *(const v8bf*)(pa + 16 + half * 8);
  v8bf b0 = *(const v8bf*)(pb + half * 8);
  v8bf b1 = *(const v8bf*)(pb + 16 + half * 8);
  v16bf r;
#pragma unroll
  for (int i = 0; i < 8; ++i) {
    float x0 = (float)a0[i] + (float)b0[i];
    float x1 = (float)a1[i] + (float)b1[i];
    r[i]     = (__bf16)fmaxf(x0, 0.0f);
    r[8 + i] = (__bf16)fmaxf(x1, 0.0f);
  }
  return r;
}

// ---------------------------------------------------------------------------
// Weight packing: 64x64 f32 (row-major, [K][N]) -> WMMA B-fragment order bf16.
// B layout (32x16 bf16): lane L: N=L%16, elems e<8 -> K=(L/16)*8+e, e>=8 -> K=16+(L/16)*8+(e-8).
// Packed index: (((kc*4 + ntile)*32 + L)*16 + e).
// ---------------------------------------------------------------------------
__global__ void __launch_bounds__(256)
pack_w_kernel(__bf16* __restrict__ dst, const float* __restrict__ srcA,
              const float* __restrict__ srcB) {
  int t = blockIdx.x * blockDim.x + threadIdx.x;
  if (t >= 4096) return;
  int e = t & 15, L = (t >> 4) & 31, frag = t >> 9;
  int kc = frag >> 2, nt = frag & 3;
  int n = nt * 16 + (L & 15);
  int k = kc * 32 + (e < 8 ? (L >> 4) * 8 + e : 16 + (L >> 4) * 8 + (e - 8));
  float v = srcA[k * 64 + n];
  if (srcB) v -= srcB[k * 64 + n];
  dst[t] = (__bf16)v;
}

// h0 = bf16(relu(BN0(x)))
__global__ void __launch_bounds__(256)
bn0_kernel(const float* __restrict__ x, const float* __restrict__ g,
           const float* __restrict__ be, const float* __restrict__ mu,
           const float* __restrict__ va, __bf16* __restrict__ h0, int NB) {
  int i = blockIdx.x * blockDim.x + threadIdx.x;
  if (i >= NB) return;
  int c = i & 63;
  float v = (x[i] - mu[c]) * (g[c] * rsqrtf(va[c] + EPSBN)) + be[c];
  h0[i] = (__bf16)fmaxf(v, 0.0f);
}

__global__ void __launch_bounds__(256)
fill_u32_kernel(unsigned* __restrict__ p, unsigned v, int n) {
  int i = blockIdx.x * blockDim.x + threadIdx.x;
  if (i < n) p[i] = v;
}

// Per-mask in-degree counts (once per call).
__global__ void __launch_bounds__(256)
count_kernel(const int* __restrict__ dst, const float* __restrict__ ea,
             int* __restrict__ c0, int* __restrict__ c1, int* __restrict__ c2,
             int E) {
  int e = blockIdx.x * blockDim.x + threadIdx.x;
  if (e >= E) return;
  int d = dst[e];
  float a = ea[e];
  if (a <= 0.0f) atomicAdd(c0 + d, 1);
  if (a >= 0.0f) atomicAdd(c1 + d, 1);
  atomicAdd(c2 + d, 1);
}

// EdgeConv node precompute: A = h0 @ (W1a-W1b) + b1 ; B = h0 @ W1b  (bf16 out)
// 32 rows per wave (two 16-row subtiles reuse each weight fragment).
__global__ void __launch_bounds__(256)
node_ab_kernel(const __bf16* __restrict__ h0, const __bf16* __restrict__ wA,
               const __bf16* __restrict__ wB, const float* __restrict__ b1,
               __bf16* __restrict__ Ab, __bf16* __restrict__ Bb, int Nn) {
  const int lane = threadIdx.x & 31;
  const int wv = blockIdx.x * (blockDim.x >> 5) + (threadIdx.x >> 5);
  const int base = wv * 32;
  if (base >= Nn) return;
  const int m = lane & 15, half = lane >> 4;
  v16bf a0[2], a1[2];
#pragma unroll
  for (int s = 0; s < 2; ++s) {
    const int row = min(base + s * 16 + m, Nn - 1);
    a0[s] = load_afrag(h0, row, 0, lane);
    a1[s] = load_afrag(h0, row, 1, lane);
  }
#pragma unroll
  for (int t = 0; t < 4; ++t) {
    v16bf wa0 = *(const v16bf*)(wA + ((0 * 4 + t) * 32 + lane) * 16);
    v16bf wa1 = *(const v16bf*)(wA + ((1 * 4 + t) * 32 + lane) * 16);
    v16bf wb0 = *(const v16bf*)(wB + ((0 * 4 + t) * 32 + lane) * 16);
    v16bf wb1 = *(const v16bf*)(wB + ((1 * 4 + t) * 32 + lane) * 16);
    const int n = t * 16 + m;
    const float bias = b1[n];
#pragma unroll
    for (int s = 0; s < 2; ++s) {
      v8f accA = {}, accB = {};
      accA = wmma_bf16(a0[s], wa0, accA);
      accA = wmma_bf16(a1[s], wa1, accA);
      accB = wmma_bf16(a0[s], wb0, accB);
      accB = wmma_bf16(a1[s], wb1, accB);
      const int nb = base + s * 16;
      if (nb + 16 <= Nn) {  // full tile fast path (uniform branch)
#pragma unroll
        for (int r = 0; r < 8; ++r) {
          const size_t o = (size_t)(nb + r + 8 * half) * 64 + n;
          Ab[o] = (__bf16)(accA[r] + bias);
          Bb[o] = (__bf16)accB[r];
        }
      } else {
#pragma unroll
        for (int r = 0; r < 8; ++r) {
          const int node = nb + r + 8 * half;
          if (node < Nn) {
            const size_t o = (size_t)node * 64 + n;
            Ab[o] = (__bf16)(accA[r] + bias);
            Bb[o] = (__bf16)accB[r];
          }
        }
      }
    }
  }
}

// Per-edge MLP layer2 + masked segment max (uint-key atomicMax).
// 32 edges per wave; weight fragments loaded once per wave.
__global__ void __launch_bounds__(256)
edge_mlp_kernel(const __bf16* __restrict__ Abuf, const __bf16* __restrict__ Bbuf,
                const __bf16* __restrict__ w2, const float* __restrict__ b2,
                const int* __restrict__ src, const int* __restrict__ dst,
                const float* __restrict__ ea, unsigned* __restrict__ aggkey,
                int E, int mode) {
  const int lane = threadIdx.x & 31;
  const int wv = blockIdx.x * (blockDim.x >> 5) + (threadIdx.x >> 5);
  const int base = wv * 32;
  if (base >= E) return;
  const int m = lane & 15, half = lane >> 4;

  int dN[2];
  unsigned mb[2];
  v16bf f0[2], f1[2];
#pragma unroll
  for (int s = 0; s < 2; ++s) {
    const int e = base + s * 16 + m;
    const bool inr = e < E;
    int sn = 0, dn = 0;
    float a = 1.0f;
    if (inr) { sn = src[e]; dn = dst[e]; a = ea[e]; }
    bool valid = inr && (mode == 2 || (mode == 0 ? (a <= 0.0f) : (a >= 0.0f)));
    mb[s] = (unsigned)__ballot(valid ? 1 : 0);
    dN[s] = dn;
    f0[s] = edge_frag(Abuf, Bbuf, dn, sn, 0, lane);
    f1[s] = edge_frag(Abuf, Bbuf, dn, sn, 1, lane);
  }
  // dst node of each output row this lane owns (rows r+8*half); t-invariant.
  int ddn[2][8];
#pragma unroll
  for (int s = 0; s < 2; ++s)
#pragma unroll
    for (int r = 0; r < 8; ++r) ddn[s][r] = __shfl(dN[s], r + 8 * half, 32);

#pragma unroll
  for (int t = 0; t < 4; ++t) {
    v16bf w0 = *(const v16bf*)(w2 + ((0 * 4 + t) * 32 + lane) * 16);
    v16bf w1 = *(const v16bf*)(w2 + ((1 * 4 + t) * 32 + lane) * 16);
    const int n = t * 16 + m;
    const float bias = b2[n];
#pragma unroll
    for (int s = 0; s < 2; ++s) {
      v8f acc = {};
      acc = wmma_bf16(f0[s], w0, acc);
      acc = wmma_bf16(f1[s], w1, acc);
#pragma unroll
      for (int r = 0; r < 8; ++r) {
        if ((mb[s] >> (r + 8 * half)) & 1u)
          atomicMax(aggkey + (size_t)ddn[s][r] * 64 + n, fkey(acc[r] + bias));
      }
    }
  }
}

// Decode max keys; empty segments -> 0; then BN+ReLU; write f32 + bf16 copies.
__global__ void __launch_bounds__(256)
h1_kernel(const unsigned* __restrict__ key, const float* __restrict__ g,
          const float* __restrict__ be, const float* __restrict__ mu,
          const float* __restrict__ va, float* __restrict__ hf,
          __bf16* __restrict__ hb, int NB) {
  int i = blockIdx.x * blockDim.x + threadIdx.x;
  if (i >= NB) return;
  int c = i & 63;
  float v = dekey(key[i]);
  if (!(v >= -3.0e38f && v <= 3.0e38f)) v = 0.0f;  // -inf (empty) -> 0
  v = (v - mu[c]) * (g[c] * rsqrtf(va[c] + EPSBN)) + be[c];
  v = fmaxf(v, 0.0f);
  hf[i] = v;
  hb[i] = (__bf16)v;
}

// Masked scatter-add of h[src] into s[dst]; float4 gather + 4 f32 atomics.
__global__ void __launch_bounds__(256)
scatter_kernel(const float* __restrict__ h, const int* __restrict__ src,
               const int* __restrict__ dst, const float* __restrict__ ea,
               float* __restrict__ s, int E, int mode) {
  int i = blockIdx.x * blockDim.x + threadIdx.x;
  if (i >= E * 16) return;
  int e = i >> 4, c = (i & 15) * 4;
  float a = ea[e];
  bool ok = (mode == 2) || (mode == 0 ? (a <= 0.0f) : (a >= 0.0f));
  if (ok) {
    const float4 v = *(const float4*)(h + (size_t)src[e] * 64 + c);
    float* sp = s + (size_t)dst[e] * 64 + c;
    atomicAdd(sp + 0, v.x);
    atomicAdd(sp + 1, v.y);
    atomicAdd(sp + 2, v.z);
    atomicAdd(sp + 3, v.w);
  }
}

__global__ void __launch_bounds__(256)
mean_kernel(const float* __restrict__ s, const int* __restrict__ cnt,
            __bf16* __restrict__ mbf, int NB) {
  int i = blockIdx.x * blockDim.x + threadIdx.x;
  if (i >= NB) return;
  float c = (float)max(cnt[i >> 6], 1);
  mbf[i] = (__bf16)(s[i] / c);
}

// SAGE: res = mean@Wl + h@Wr + b, then BN epilogue. 32 rows per wave.
// MODE 0: hnext = relu(BN(res)) -> write f32 + bf16 ; MODE 1: out += BN(res)
template <int MODE>
__global__ void __launch_bounds__(256)
sage_gemm_kernel(const __bf16* __restrict__ meanbf, const __bf16* __restrict__ hbf,
                 const __bf16* __restrict__ wl, const __bf16* __restrict__ wr,
                 const float* __restrict__ bias, const float* __restrict__ g,
                 const float* __restrict__ be, const float* __restrict__ mu,
                 const float* __restrict__ va, float* __restrict__ outf,
                 __bf16* __restrict__ outb, int Nn) {
  const int lane = threadIdx.x & 31;
  const int wv = blockIdx.x * (blockDim.x >> 5) + (threadIdx.x >> 5);
  const int base = wv * 32;
  if (base >= Nn) return;
  const int m = lane & 15, half = lane >> 4;
  v16bf mA0[2], mA1[2], hA0[2], hA1[2];
#pragma unroll
  for (int s = 0; s < 2; ++s) {
    const int row = min(base + s * 16 + m, Nn - 1);
    mA0[s] = load_afrag(meanbf, row, 0, lane);
    mA1[s] = load_afrag(meanbf, row, 1, lane);
    hA0[s] = load_afrag(hbf, row, 0, lane);
    hA1[s] = load_afrag(hbf, row, 1, lane);
  }
#pragma unroll
  for (int t = 0; t < 4; ++t) {
    v16bf wl0 = *(const v16bf*)(wl + ((0 * 4 + t) * 32 + lane) * 16);
    v16bf wl1 = *(const v16bf*)(wl + ((1 * 4 + t) * 32 + lane) * 16);
    v16bf wr0 = *(const v16bf*)(wr + ((0 * 4 + t) * 32 + lane) * 16);
    v16bf wr1 = *(const v16bf*)(wr + ((1 * 4 + t) * 32 + lane) * 16);
    const int n = t * 16 + m;
    const float sc = g[n] * rsqrtf(va[n] + EPSBN);
    const float sh = be[n], mm = mu[n], bi = bias[n];
#pragma unroll
    for (int s = 0; s < 2; ++s) {
      v8f acc = {};
      acc = wmma_bf16(mA0[s], wl0, acc);
      acc = wmma_bf16(mA1[s], wl1, acc);
      acc = wmma_bf16(hA0[s], wr0, acc);
      acc = wmma_bf16(hA1[s], wr1, acc);
      const int nb = base + s * 16;
      if (nb + 16 <= Nn) {  // full tile fast path (uniform branch)
#pragma unroll
        for (int r = 0; r < 8; ++r) {
          const size_t o = (size_t)(nb + r + 8 * half) * 64 + n;
          float v = (acc[r] + bi - mm) * sc + sh;
          if (MODE == 0) {
            v = fmaxf(v, 0.0f);
            outf[o] = v;
            outb[o] = (__bf16)v;
          } else {
            outf[o] += v;
          }
        }
      } else {
#pragma unroll
        for (int r = 0; r < 8; ++r) {
          const int node = nb + r + 8 * half;
          if (node < Nn) {
            const size_t o = (size_t)node * 64 + n;
            float v = (acc[r] + bi - mm) * sc + sh;
            if (MODE == 0) {
              v = fmaxf(v, 0.0f);
              outf[o] = v;
              outb[o] = (__bf16)v;
            } else {
              outf[o] += v;
            }
          }
        }
      }
    }
  }
}

// ---------------------------------------------------------------------------
// Host side
// ---------------------------------------------------------------------------
extern "C" void kernel_launch(void* const* d_in, const int* in_sizes, int n_in,
                              void* d_out, int out_size, void* d_ws, size_t ws_size,
                              hipStream_t stream) {
  const float* x = (const float*)d_in[0];
  const int* ei = (const int*)d_in[1];
  const float* ea = (const float*)d_in[2];
  const int Nn = in_sizes[0] / 64;
  const int E = in_sizes[2];
  const int NB = Nn * 64;
  const int* srcI = ei;
  const int* dstI = ei + E;

  auto P = [&](int i) { return (const float*)d_in[i]; };

  // Workspace carve (256B aligned).
  char* w = (char*)d_ws;
  size_t cur = 0;
  auto carve = [&](size_t bytes) {
    void* p = w + cur;
    cur += (bytes + 255) & ~(size_t)255;
    return p;
  };
  __bf16* packW = (__bf16*)carve((size_t)3 * 7 * 4096 * 2);
  __bf16* h0bf = (__bf16*)carve((size_t)NB * 2);
  __bf16* Abuf = (__bf16*)carve((size_t)NB * 2);
  __bf16* Bbuf = (__bf16*)carve((size_t)NB * 2);
  unsigned* aggkey = (unsigned*)carve((size_t)NB * 4);
  float* h1f = (float*)carve((size_t)NB * 4);
  __bf16* h1bf = (__bf16*)carve((size_t)NB * 2);
  float* sbuf = (float*)carve((size_t)NB * 4);
  __bf16* meanbf = (__bf16*)carve((size_t)NB * 2);
  float* h2f = (float*)carve((size_t)NB * 4);
  __bf16* h2bf = (__bf16*)carve((size_t)NB * 2);
  int* cnt0 = (int*)carve((size_t)Nn * 4);
  int* cnt1 = (int*)carve((size_t)Nn * 4);
  int* cnt2 = (int*)carve((size_t)Nn * 4);
  int* cntArr[3] = {cnt0, cnt1, cnt2};

  const dim3 B256(256);
  const int gNB = (NB + 255) / 256;
  const int gE = (E + 255) / 256;
  const int gE16 = (E * 16 + 255) / 256;
  const int nT32 = (Nn + 31) / 32, gNodeW = (nT32 + 7) / 8;
  const int eT32 = (E + 31) / 32, gEdgeW = (eT32 + 7) / 8;

  // Zero the output accumulator and degree counters each call (deterministic).
  hipMemsetAsync(d_out, 0, (size_t)NB * 4, stream);
  hipMemsetAsync(cnt0, 0, (size_t)Nn * 4 * 3, stream);

  // Pack all branch weights into WMMA B-fragment order (bf16).
  for (int b = 0; b < 3; ++b) {
    const int pb = 7 + b * 22;
    __bf16* pw = packW + (size_t)b * 7 * 4096;
    const float* W1 = P(pb + 0);  // (128,64): rows 0..63 = W1a (xi), 64..127 = W1b
    pack_w_kernel<<<16, B256, 0, stream>>>(pw + 0 * 4096, W1, W1 + 4096); // W1a-W1b
    pack_w_kernel<<<16, B256, 0, stream>>>(pw + 1 * 4096, W1 + 4096, nullptr); // W1b
    pack_w_kernel<<<16, B256, 0, stream>>>(pw + 2 * 4096, P(pb + 2), nullptr);  // W2
    pack_w_kernel<<<16, B256, 0, stream>>>(pw + 3 * 4096, P(pb + 8), nullptr);  // sg2.Wl
    pack_w_kernel<<<16, B256, 0, stream>>>(pw + 4 * 4096, P(pb + 9), nullptr);  // sg2.Wr
    pack_w_kernel<<<16, B256, 0, stream>>>(pw + 5 * 4096, P(pb + 15), nullptr); // sg3.Wl
    pack_w_kernel<<<16, B256, 0, stream>>>(pw + 6 * 4096, P(pb + 16), nullptr); // sg3.Wr
  }

  // h0 = bf16(relu(BN0(x)))
  bn0_kernel<<<gNB, B256, 0, stream>>>(x, P(3), P(4), P(5), P(6), h0bf, NB);
  // in-degree counts per mask
  count_kernel<<<gE, B256, 0, stream>>>(dstI, ea, cnt0, cnt1, cnt2, E);

  for (int b = 0; b < 3; ++b) {
    const int pb = 7 + b * 22;
    const int mode = b;  // 0: ea<=0, 1: ea>=0, 2: all
    __bf16* pw = packW + (size_t)b * 7 * 4096;

    // EdgeConv node precompute + masked edge GEMM + segment max
    node_ab_kernel<<<gNodeW, B256, 0, stream>>>(h0bf, pw + 0 * 4096, pw + 1 * 4096,
                                                P(pb + 1), Abuf, Bbuf, Nn);
    fill_u32_kernel<<<gNB, B256, 0, stream>>>(aggkey, NEGINF_KEY, NB);
    edge_mlp_kernel<<<gEdgeW, B256, 0, stream>>>(Abuf, Bbuf, pw + 2 * 4096, P(pb + 3),
                                                 srcI, dstI, ea, aggkey, E, mode);
    h1_kernel<<<gNB, B256, 0, stream>>>(aggkey, P(pb + 4), P(pb + 5), P(pb + 6),
                                        P(pb + 7), h1f, h1bf, NB);

    // SAGE layer 2 (mean aggr) -> relu(BN2(.))
    hipMemsetAsync(sbuf, 0, (size_t)NB * 4, stream);
    scatter_kernel<<<gE16, B256, 0, stream>>>(h1f, srcI, dstI, ea, sbuf, E, mode);
    mean_kernel<<<gNB, B256, 0, stream>>>(sbuf, cntArr[mode], meanbf, NB);
    sage_gemm_kernel<0><<<gNodeW, B256, 0, stream>>>(
        meanbf, h1bf, pw + 3 * 4096, pw + 4 * 4096, P(pb + 10), P(pb + 11),
        P(pb + 12), P(pb + 13), P(pb + 14), h2f, h2bf, Nn);

    // SAGE layer 3 -> out += BN3(.)
    hipMemsetAsync(sbuf, 0, (size_t)NB * 4, stream);
    scatter_kernel<<<gE16, B256, 0, stream>>>(h2f, srcI, dstI, ea, sbuf, E, mode);
    mean_kernel<<<gNB, B256, 0, stream>>>(sbuf, cntArr[mode], meanbf, NB);
    sage_gemm_kernel<1><<<gNodeW, B256, 0, stream>>>(
        meanbf, h2bf, pw + 5 * 4096, pw + 6 * 4096, P(pb + 17), P(pb + 18),
        P(pb + 19), P(pb + 20), P(pb + 21), (float*)d_out, nullptr, Nn);
  }
}
